// MultiHeadAttention_40295383171462
// MI455X (gfx1250) — matrix operand order repaired and checked by reading, not compiled, not run on hardware
//
#include <hip/hip_runtime.h>
#include <hip/hip_bf16.h>
#include <cstdint>
#include <cstddef>

// ---------------------------------------------------------------------------
// MultiHeadAttention (N=4096, D=1024, H=16, d_k=64) for gfx1250 (MI455X).
// Flash-attention structure, bf16 WMMA (v_wmma_f32_16x16x32_bf16) everywhere,
// double-buffered K/V staging via GLOBAL_LOAD_ASYNC_TO_LDS_B128 (ASYNCcnt).
// Workspace layout (28 MiB): Xb[4096*1024] bf16 | Wb[2048*1024] bf16 |
//                            proj[4096*2048] bf16
// ---------------------------------------------------------------------------

typedef __attribute__((ext_vector_type(16))) __bf16 v16bf;
typedef __attribute__((ext_vector_type(8)))  float  v8f;

union Frag {
  v16bf          v;
  uint4          q[2];
  unsigned short u[16];
};

__device__ __forceinline__ unsigned short f2bf(float f) {
  unsigned int u = __builtin_bit_cast(unsigned int, f);
  u += 0x7fffu + ((u >> 16) & 1u);          // round-to-nearest-even
  return (unsigned short)(u >> 16);
}

// Async global -> LDS copy (16B per lane). Portable inline-asm form
// (the clang builtin arity differs between toolchains).
__device__ __forceinline__ void async_load_b128(const void* gaddr, void* lds) {
  unsigned int l = (unsigned int)(uintptr_t)lds;   // low 32b of generic = LDS offset
  asm volatile("global_load_async_to_lds_b128 %0, %1, off"
               :: "v"(l), "v"(gaddr) : "memory");
}

constexpr int N_     = 4096;
constexpr int D_     = 1024;
constexpr int H_     = 16;
constexpr int DK     = 64;
constexpr int PROJ_W = 2048;

// -------------------------- f32 -> bf16 conversion -------------------------
__global__ void cvt_f32_bf16(const float* __restrict__ src,
                             unsigned short* __restrict__ dst, int n) {
  int i = blockIdx.x * blockDim.x + threadIdx.x;
  if (i < n) dst[i] = f2bf(src[i]);
}

// -------------------- proj = Xb @ Wb^T  (bf16 in, bf16 out) ----------------
// grid = (64 m-groups, 32 n-groups), block = 128 (4 waves).
// wave w: rows m0 = (bx*4+w)*16, cols n0 = by*64 (4 x 16-wide WMMA tiles).
__global__ __launch_bounds__(128)
void qkv_gemm(const unsigned short* __restrict__ Xb,
              const unsigned short* __restrict__ Wb,
              unsigned short* __restrict__ proj) {
  const int lane = threadIdx.x & 31;
  const int wave = threadIdx.x >> 5;
  const int lr   = lane & 15;
  const int lh   = lane >> 4;                       // 0 / 1 (lane half)
  const int m0   = (blockIdx.x * 4 + wave) * 16;
  const int n0   = blockIdx.y * 64;

  const unsigned short* arow = Xb + (size_t)(m0 + lr) * D_;

  v8f acc[4];
#pragma unroll
  for (int t = 0; t < 4; ++t) acc[t] = {};

  for (int kb = 0; kb < D_; kb += 32) {
    // A-frag (16x32 bf16): lane half selects K sub-octet per ISA layout.
    Frag a;
    a.q[0] = *(const uint4*)(arow + kb + lh * 8);
    a.q[1] = *(const uint4*)(arow + kb + 16 + lh * 8);
#pragma unroll
    for (int t = 0; t < 4; ++t) {
      // B-frag (32x16): lane holds column n = n0+16t+lr = one W row (contiguous K).
      const unsigned short* bp =
          Wb + (size_t)(n0 + 16 * t + lr) * D_ + kb + lh * 8;
      Frag b;
      b.q[0] = *(const uint4*)(bp);
      b.q[1] = *(const uint4*)(bp + 16);
      acc[t] = __builtin_amdgcn_wmma_f32_16x16x32_bf16(
          false, a.v, false, b.v, (short)0, acc[t], false, false);
    }
  }

#pragma unroll
  for (int t = 0; t < 4; ++t) {
    const int col = n0 + 16 * t + lr;
#pragma unroll
    for (int r = 0; r < 8; ++r) {
      const int row = m0 + r + 8 * lh;              // C layout: M = r + 8*half
      proj[(size_t)row * PROJ_W + col] = f2bf(acc[t][r]);
    }
  }
}

// ------------------------------ flash attention ----------------------------
// grid = (N/64 q-blocks, H heads), block = 128 (4 waves, 16 q-rows each).
// Double-buffered async K/V staging: while block i is consumed by WMMAs,
// block i+1 streams into the other LDS buffer via ASYNC loads.
__global__ __launch_bounds__(128)
void attn(const unsigned short* __restrict__ proj, float* __restrict__ out) {
  constexpr int LP = 72;                            // padded LDS row stride
  __shared__ __align__(16) unsigned short Ks[2][64 * LP];  // K blocks, row-major
  __shared__ __align__(16) unsigned short Vs[2][64 * LP];  // V blocks, row-major
  __shared__ __align__(16) unsigned short Vt[64 * LP];     // V transposed [d][key]
  __shared__ __align__(16) unsigned short Ps[4][16 * LP];  // per-wave P staging

  const int lane = threadIdx.x & 31;
  const int wave = threadIdx.x >> 5;
  const int lr   = lane & 15;
  const int lh   = lane >> 4;
  const int head = blockIdx.y;
  const int m0   = blockIdx.x * 64 + wave * 16;

  const int tid  = threadIdx.x;
  const int krow = tid >> 1;                        // staging row per thread
  const int c0   = (tid & 1) * 32;                  // staging col half

  const unsigned short* qbase = proj + head * DK;           // + row*PROJ_W
  const unsigned short* vbase = proj + D_ + head * DK;

  // Q A-frags (16 rows x 64 d), two 16x32 fragments.
  Frag qa[2];
  {
    const unsigned short* qrow = qbase + (size_t)(m0 + lr) * PROJ_W;
#pragma unroll
    for (int f = 0; f < 2; ++f) {
      qa[f].q[0] = *(const uint4*)(qrow + 32 * f + lh * 16);
      qa[f].q[1] = *(const uint4*)(qrow + 32 * f + lh * 16 + 8);
    }
  }

  v8f o[4];
#pragma unroll
  for (int t = 0; t < 4; ++t) o[t] = {};
  float mrow[8], lrow[8];
#pragma unroll
  for (int r = 0; r < 8; ++r) { mrow[r] = -3.0e38f; lrow[r] = 0.0f; }

  const float scale = 0.125f;                       // 1/sqrt(64)

  // Issue async staging of block `kb` into buffer `buf` (8 x b128 per thread).
  auto issue_block = [&](int kb, int buf) {
    const unsigned short* kg = qbase + (size_t)(kb + krow) * PROJ_W + c0;
    const unsigned short* vg = vbase + (size_t)(kb + krow) * PROJ_W + c0;
    unsigned short* kl = &Ks[buf][krow * LP + c0];
    unsigned short* vl = &Vs[buf][krow * LP + c0];
#pragma unroll
    for (int j = 0; j < 4; ++j) {
      async_load_b128(kg + 8 * j, kl + 8 * j);
      async_load_b128(vg + 8 * j, vl + 8 * j);
    }
  };

  issue_block(0, 0);                                // prologue

  for (int kb = 0; kb < N_; kb += 64) {
    const int buf = (kb >> 6) & 1;

    // Prefetch next block into the other buffer, then wait for THIS block.
    // Async loads complete in order, so <=8 outstanding => older 8 are done.
    if (kb + 64 < N_) {
      issue_block(kb + 64, buf ^ 1);
      asm volatile("s_wait_asynccnt 8" ::: "memory");
    } else {
      asm volatile("s_wait_asynccnt 0" ::: "memory");
    }
    __syncthreads();                                // block data visible to all

    // ---- LDS->LDS transpose of V (row-major -> Vt[d][key]) ----
    {
      uint4 tmp[4];
#pragma unroll
      for (int j = 0; j < 4; ++j)
        tmp[j] = *(const uint4*)&Vs[buf][krow * LP + c0 + 8 * j];
      const unsigned short* tp = (const unsigned short*)tmp;
#pragma unroll
      for (int j = 0; j < 32; ++j)
        Vt[(c0 + j) * LP + krow] = tp[j];
    }
    __syncthreads();

    // ---- S = Q K^T : 4 tiles of 16x16, 2 WMMA (K=32) each ----
    v8f s[4];
#pragma unroll
    for (int t = 0; t < 4; ++t) {
      const unsigned short* kp = &Ks[buf][(16 * t + lr) * LP + lh * 16];
      Frag b0, b1;
      b0.q[0] = *(const uint4*)(kp);
      b0.q[1] = *(const uint4*)(kp + 8);
      b1.q[0] = *(const uint4*)(kp + 32);
      b1.q[1] = *(const uint4*)(kp + 40);
      v8f c = {};
      c = __builtin_amdgcn_wmma_f32_16x16x32_bf16(false, qa[0].v, false, b0.v,
                                                  (short)0, c, false, false);
      c = __builtin_amdgcn_wmma_f32_16x16x32_bf16(false, qa[1].v, false, b1.v,
                                                  (short)0, c, false, false);
      s[t] = c;
    }

    // ---- online softmax (row stats per lane; 16-lane butterfly reduce) ----
    float mnew[8], alpha[8];
#pragma unroll
    for (int r = 0; r < 8; ++r) {
      float mx = fmaxf(fmaxf(s[0][r], s[1][r]), fmaxf(s[2][r], s[3][r]));
#pragma unroll
      for (int d = 1; d < 16; d <<= 1) mx = fmaxf(mx, __shfl_xor(mx, d, 16));
      mx *= scale;
      mnew[r]  = fmaxf(mrow[r], mx);
      alpha[r] = __expf(mrow[r] - mnew[r]);
      mrow[r]  = mnew[r];
    }
#pragma unroll
    for (int t = 0; t < 4; ++t)
#pragma unroll
      for (int r = 0; r < 8; ++r)
        s[t][r] = __expf(s[t][r] * scale - mnew[r]);
#pragma unroll
    for (int r = 0; r < 8; ++r) {
      float ps = s[0][r] + s[1][r] + s[2][r] + s[3][r];
#pragma unroll
      for (int d = 1; d < 16; d <<= 1) ps += __shfl_xor(ps, d, 16);
      lrow[r] = lrow[r] * alpha[r] + ps;
#pragma unroll
      for (int t = 0; t < 4; ++t) o[t][r] *= alpha[r];
    }

    // ---- restage P through per-wave LDS to get A-layout ----
    unsigned short* pw = Ps[wave];
#pragma unroll
    for (int t = 0; t < 4; ++t)
#pragma unroll
      for (int r = 0; r < 8; ++r)
        pw[(r + 8 * lh) * LP + 16 * t + lr] = f2bf(s[t][r]);
    asm volatile("s_wait_dscnt 0" ::: "memory");    // wave-internal LDS WAR/RAW

    Frag pa[2];
    {
      const unsigned short* pr = pw + lr * LP;
#pragma unroll
      for (int f = 0; f < 2; ++f) {
        pa[f].q[0] = *(const uint4*)(pr + 32 * f + lh * 16);
        pa[f].q[1] = *(const uint4*)(pr + 32 * f + lh * 16 + 8);
      }
    }

    // ---- O += P V : 4 output tiles, 2 WMMA (K=32) each ----
#pragma unroll
    for (int t = 0; t < 4; ++t) {
      const unsigned short* vp = &Vt[(16 * t + lr) * LP + lh * 16];
      Frag vb0, vb1;
      vb0.q[0] = *(const uint4*)(vp);
      vb0.q[1] = *(const uint4*)(vp + 8);
      vb1.q[0] = *(const uint4*)(vp + 32);
      vb1.q[1] = *(const uint4*)(vp + 40);
      o[t] = __builtin_amdgcn_wmma_f32_16x16x32_bf16(false, pa[0].v, false,
                                                     vb0.v, (short)0, o[t],
                                                     false, false);
      o[t] = __builtin_amdgcn_wmma_f32_16x16x32_bf16(false, pa[1].v, false,
                                                     vb1.v, (short)0, o[t],
                                                     false, false);
    }
    __syncthreads();                                // protect LDS for next iter
  }

  // ---- finalize: ctx = O * (1/l) ----
#pragma unroll
  for (int t = 0; t < 4; ++t) {
#pragma unroll
    for (int r = 0; r < 8; ++r) {
      const float rinv = __builtin_amdgcn_rcpf(lrow[r]);
      const int row = m0 + r + 8 * lh;
      const int col = head * DK + 16 * t + lr;
      out[(size_t)row * D_ + col] = o[t][r] * rinv;
    }
  }
}

// ------------------------------- launcher ----------------------------------
extern "C" void kernel_launch(void* const* d_in, const int* in_sizes, int n_in,
                              void* d_out, int out_size, void* d_ws,
                              size_t ws_size, hipStream_t stream) {
  const float* X = (const float*)d_in[0];   // [4096,1024]
  const float* W = (const float*)d_in[1];   // [2048,1024]
  float* out = (float*)d_out;               // [4096,1024]

  unsigned short* Xb   = (unsigned short*)d_ws;                 // 8 MiB
  unsigned short* Wb   = Xb + (size_t)N_ * D_;                  // 4 MiB
  unsigned short* proj = Wb + (size_t)PROJ_W * D_;              // 16 MiB

  const int nX = N_ * D_;
  const int nW = PROJ_W * D_;
  cvt_f32_bf16<<<nX / 256, 256, 0, stream>>>(X, Xb, nX);
  cvt_f32_bf16<<<nW / 256, 256, 0, stream>>>(W, Wb, nW);

  qkv_gemm<<<dim3(64, 32), 128, 0, stream>>>(Xb, Wb, proj);

  attn<<<dim3(N_ / 64, H_), 128, 0, stream>>>(proj, out);
}
